// GCMC_73727408603557
// MI455X (gfx1250) — matrix operand order repaired and checked by reading, not compiled, not run on hardware
//
#include <hip/hip_runtime.h>
#include <hip/hip_fp16.h>

#define NU_ 6000
#define NV_ 4000
#define FD_ 1024
#define HD_ 64
#define RR_ 5
#define SD_ 128
#define SHD_ 64
#define OD_ 64
#define NE_ 200000
#define CAT_ (RR_ * HD_ + SHD_)   // 384

typedef _Float16 h16;
typedef __attribute__((ext_vector_type(2)))  _Float16 v2h;
typedef __attribute__((ext_vector_type(16))) _Float16 v16h;
typedef __attribute__((ext_vector_type(8)))  float    v8f;

#define WMMA_F16(A, B, C) \
  __builtin_amdgcn_wmma_f32_16x16x32_f16(false, (A), false, (B), (short)0, (C), false, false)

// ---------------- WMMA fragment helpers (CDNA5 16x16x32 f16, wave32) -------

// A-fragment: 16x32 f16, row-major source, lda in elements.
// Lane l holds row m = m0 + (l&15); VGPR j holds K = k0 + 2*(j&3) + 16*(j>=4) + 8*(l>=16).
static __device__ __forceinline__ v16h load_a_f16(const h16* __restrict__ A, int lda,
                                                  int m0, int k0) {
  const int lane = threadIdx.x & 31;
  const h16* p = A + (size_t)(m0 + (lane & 15)) * lda + (k0 + ((lane & 16) ? 8 : 0));
  v16h a;
#pragma unroll
  for (int j = 0; j < 8; ++j) {
    const int koff = 2 * (j & 3) + ((j & 4) ? 16 : 0);
    v2h pr = *(const v2h*)(p + koff);
    a[2 * j]     = pr[0];
    a[2 * j + 1] = pr[1];
  }
  return a;
}

// Same as above, but f32 source with ReLU + convert at load (for concat hidden).
static __device__ __forceinline__ v16h load_a_f32_relu(const float* __restrict__ A, int lda,
                                                       int m0, int k0) {
  const int lane = threadIdx.x & 31;
  const float* p = A + (size_t)(m0 + (lane & 15)) * lda + (k0 + ((lane & 16) ? 8 : 0));
  v16h a;
#pragma unroll
  for (int j = 0; j < 8; ++j) {
    const int koff = 2 * (j & 3) + ((j & 4) ? 16 : 0);
    a[2 * j]     = (h16)fmaxf(p[koff],     0.f);
    a[2 * j + 1] = (h16)fmaxf(p[koff + 1], 0.f);
  }
  return a;
}

// B-fragment: 32x16 f16 where B[k][n] = Bt[n][k] (Bt row-major [N][K], ldb in elems).
// Lane l holds col n = n0 + (l&15); half i holds K = k0 + 8*(l>=16) + (i&7) + 16*(i>=8)  -> matches A.
static __device__ __forceinline__ v16h load_bT(const h16* __restrict__ Bt, int ldb,
                                               int n0, int k0) {
  const int lane = threadIdx.x & 31;
  const h16* p = Bt + (size_t)(n0 + (lane & 15)) * ldb + (k0 + ((lane & 16) ? 8 : 0));
  typedef __attribute__((ext_vector_type(8))) _Float16 v8h_;
  v8h_ lo = *(const v8h_*)p;   // 16B-aligned at every call site
  v8h_ hi = *(const v8h_*)(p + 16);
  return __builtin_shufflevector(lo, hi, 0, 1, 2, 3, 4, 5, 6, 7, 8, 9, 10, 11, 12, 13, 14, 15);
}

// C/D store: VGPR v holds row m0 + v + 8*(lane>=16), col = n0 + (lane&15).
static __device__ __forceinline__ void store_c(float* __restrict__ C, int ldc,
                                               int m0, int n0, v8f c) {
  const int lane = threadIdx.x & 31;
  const int col  = n0 + (lane & 15);
  const int mb   = m0 + ((lane & 16) ? 8 : 0);
#pragma unroll
  for (int v = 0; v < 8; ++v)
    C[(size_t)(mb + v) * ldc + col] = c[v];
}

// ---------------- stage 0: conversions / transposes -------------------------

__global__ void cvt_f16_kernel(const float* __restrict__ src, h16* __restrict__ dst,
                               size_t n) {
  size_t t = (size_t)blockIdx.x * blockDim.x + threadIdx.x;
  if (t < n) dst[t] = (h16)src[t];
}

// W [R,1024,64] f32 -> Wt [R,64,1024] f16
__global__ void transpose_w_kernel(const float* __restrict__ W, h16* __restrict__ Wt) {
  int t = blockIdx.x * blockDim.x + threadIdx.x;
  if (t >= RR_ * HD_ * FD_) return;
  int k = t % FD_;
  int n = (t / FD_) % HD_;
  int r = t / (FD_ * HD_);
  Wt[t] = (h16)W[((size_t)r * FD_ + k) * HD_ + n];
}

// Wc [384,64] f32 -> Wct [64,384] f16
__global__ void transpose_cat_kernel(const float* __restrict__ Wc, h16* __restrict__ Wct) {
  int t = blockIdx.x * blockDim.x + threadIdx.x;
  if (t >= OD_ * CAT_) return;
  int k = t % CAT_;
  int n = t / CAT_;
  Wct[t] = (h16)Wc[(size_t)k * OD_ + n];
}

// ---------------- stage 1: per-relation feature GEMM (WMMA) -----------------
// T[r] = F @ W[r] : [M,1024] x [1024,64] -> [M,64], one wave per 16x64 tile.
// k-loop is double-buffered: fragments for step k+1 are in flight while the
// 4 WMMAs of step k execute (avoids s_wait_loadcnt 0 before every WMMA).
__global__ void __launch_bounds__(256)
gemm_feat_kernel(const h16* __restrict__ F, const h16* __restrict__ Wt,
                 float* __restrict__ T, int M) {
  const int wave   = (blockIdx.x * blockDim.x + threadIdx.x) >> 5;
  const int mtiles = M >> 4;
  if (wave >= RR_ * mtiles) return;
  const int r  = wave / mtiles;
  const int mt = wave % mtiles;
  const h16* Wr = Wt + (size_t)r * HD_ * FD_;
  float*     Tr = T  + (size_t)r * M  * HD_;

  v8f acc[4] = {v8f{}, v8f{}, v8f{}, v8f{}};
  v16h a = load_a_f16(F, FD_, mt * 16, 0);
  v16h b[4];
#pragma unroll
  for (int nt = 0; nt < 4; ++nt) b[nt] = load_bT(Wr, FD_, nt * 16, 0);

#pragma unroll 2
  for (int k0 = 0; k0 < FD_ - 32; k0 += 32) {
    v16h an = load_a_f16(F, FD_, mt * 16, k0 + 32);
    v16h bn[4];
#pragma unroll
    for (int nt = 0; nt < 4; ++nt) bn[nt] = load_bT(Wr, FD_, nt * 16, k0 + 32);
#pragma unroll
    for (int nt = 0; nt < 4; ++nt) acc[nt] = WMMA_F16(a, b[nt], acc[nt]);
    a = an;
#pragma unroll
    for (int nt = 0; nt < 4; ++nt) b[nt] = bn[nt];
  }
#pragma unroll
  for (int nt = 0; nt < 4; ++nt) acc[nt] = WMMA_F16(a, b[nt], acc[nt]);

#pragma unroll
  for (int nt = 0; nt < 4; ++nt) store_c(Tr, HD_, mt * 16, nt * 16, acc[nt]);
}

// ---------------- stage 2: edge gather/scatter (hardware f32 atomics) -------
// hcat_u[u, r*64+h] += tv[r,v,h]*w ; hcat_v[v, r*64+h] += tu[r,u,h]*w
__global__ void scatter_kernel(const int* __restrict__ u_idx, const int* __restrict__ v_idx,
                               const float* __restrict__ ew,
                               const float* __restrict__ tu, const float* __restrict__ tv,
                               float* __restrict__ hcat_u, float* __restrict__ hcat_v) {
  long long t = (long long)blockIdx.x * blockDim.x + threadIdx.x;
  if (t >= (long long)RR_ * NE_ * HD_) return;
  const int  h  = (int)(t & 63);
  long long  eg = t >> 6;
  const int  r  = (int)(eg / NE_);
  const int  e  = (int)(eg % NE_);
  const int  u  = u_idx[(size_t)r * NE_ + e];
  const int  v  = v_idx[(size_t)r * NE_ + e];
  const float w = ew[(size_t)r * NE_ + e];
  unsafeAtomicAdd(&hcat_u[(size_t)u * CAT_ + r * HD_ + h],
                  tv[((size_t)r * NV_ + v) * HD_ + h] * w);
  unsafeAtomicAdd(&hcat_v[(size_t)v * CAT_ + r * HD_ + h],
                  tu[((size_t)r * NU_ + u) * HD_ + h] * w);
}

// ---------------- stage 3: side MLP into concat columns [320..383] ----------
__global__ void side_kernel(const float* __restrict__ Sf, const float* __restrict__ Ws,
                            const float* __restrict__ bs, float* __restrict__ hcat, int M) {
  int t = blockIdx.x * blockDim.x + threadIdx.x;
  if (t >= M * SHD_) return;
  const int o   = t & 63;
  const int row = t >> 6;
  float acc = bs[o];
#pragma unroll 8
  for (int k = 0; k < SD_; ++k)
    acc = fmaf(Sf[(size_t)row * SD_ + k], Ws[(size_t)k * SHD_ + o], acc);
  hcat[(size_t)row * CAT_ + RR_ * HD_ + o] = fmaxf(acc, 0.f);
}

// ---------------- stage 4: concat GEMM -> embed (WMMA, relu A-load + D) -----
__global__ void __launch_bounds__(256)
gemm_cat_kernel(const float* __restrict__ hcat, const h16* __restrict__ Wct,
                float* __restrict__ E, h16* __restrict__ Eh, int M) {
  const int wave = (blockIdx.x * blockDim.x + threadIdx.x) >> 5;
  if (wave >= (M >> 4)) return;
  const int mt = wave;

  v8f acc[4] = {v8f{}, v8f{}, v8f{}, v8f{}};
  v16h a = load_a_f32_relu(hcat, CAT_, mt * 16, 0);
  v16h b[4];
#pragma unroll
  for (int nt = 0; nt < 4; ++nt) b[nt] = load_bT(Wct, CAT_, nt * 16, 0);

#pragma unroll 2
  for (int k0 = 0; k0 < CAT_ - 32; k0 += 32) {
    v16h an = load_a_f32_relu(hcat, CAT_, mt * 16, k0 + 32);
    v16h bn[4];
#pragma unroll
    for (int nt = 0; nt < 4; ++nt) bn[nt] = load_bT(Wct, CAT_, nt * 16, k0 + 32);
#pragma unroll
    for (int nt = 0; nt < 4; ++nt) acc[nt] = WMMA_F16(a, b[nt], acc[nt]);
    a = an;
#pragma unroll
    for (int nt = 0; nt < 4; ++nt) b[nt] = bn[nt];
  }
#pragma unroll
  for (int nt = 0; nt < 4; ++nt) acc[nt] = WMMA_F16(a, b[nt], acc[nt]);

  const int lane = threadIdx.x & 31;
  const int mb   = mt * 16 + ((lane & 16) ? 8 : 0);
#pragma unroll
  for (int nt = 0; nt < 4; ++nt) {
    const int col = nt * 16 + (lane & 15);
#pragma unroll
    for (int v = 0; v < 8; ++v) {
      float x = fmaxf(acc[nt][v], 0.f);          // relu(embed)
      E[(size_t)(mb + v) * OD_ + col] = x;
      if (Eh) Eh[(size_t)(mb + v) * OD_ + col] = (h16)x;
    }
  }
}

// ---------------- stage 5: Ah[r] = embed_u @ Q[r]  (small, VALU f32) --------
__global__ void q_kernel(const float* __restrict__ Eu, const float* __restrict__ Q,
                         h16* __restrict__ Ah) {
  int t = blockIdx.x * blockDim.x + threadIdx.x;
  if (t >= RR_ * NU_ * OD_) return;
  const int o  = t & 63;
  const int ru = t >> 6;
  const int u  = ru % NU_;
  const int r  = ru / NU_;
  const float* q = Q  + (size_t)r * OD_ * OD_;
  const float* e = Eu + (size_t)u * OD_;
  float acc = 0.f;
#pragma unroll 8
  for (int f = 0; f < OD_; ++f) acc = fmaf(e[f], q[(size_t)f * OD_ + o], acc);
  Ah[t] = (h16)acc;   // layout [r][u][o]
}

// ---------------- stage 6: score[r] = Ah[r] @ embed_v^T (WMMA, NT stores) ---
// One wave per 16x80 output tile; K=64. All 12 fragments (2 A + 10 B) are
// loaded in one clause up front, then 10 WMMAs issue back-to-back; consecutive
// WMMAs target different accumulators so the matrix pipe never stalls on a
// dependent chain. Output streamed with non-temporal stores (480 MB, HBM-bound).
__global__ void __launch_bounds__(256)
score_kernel(const h16* __restrict__ Ah, const h16* __restrict__ Bh,
             float* __restrict__ S) {
  const int wave   = (blockIdx.x * blockDim.x + threadIdx.x) >> 5;
  const int mtiles = NU_ / 16;   // 375
  const int ntiles = NV_ / 80;   // 50
  if (wave >= RR_ * mtiles * ntiles) return;
  const int r   = wave / (mtiles * ntiles);
  const int rem = wave % (mtiles * ntiles);
  const int mt  = rem / ntiles;
  const int nt  = rem % ntiles;
  const h16* Ar = Ah + (size_t)r * NU_ * OD_;
  float*     Sr = S  + (size_t)r * NU_ * NV_;

  v16h a0 = load_a_f16(Ar, OD_, mt * 16, 0);
  v16h a1 = load_a_f16(Ar, OD_, mt * 16, 32);
  v16h b0[5], b1[5];
#pragma unroll
  for (int j = 0; j < 5; ++j) {
    b0[j] = load_bT(Bh, OD_, nt * 80 + j * 16, 0);
    b1[j] = load_bT(Bh, OD_, nt * 80 + j * 16, 32);
  }

  v8f acc[5] = {v8f{}, v8f{}, v8f{}, v8f{}, v8f{}};
#pragma unroll
  for (int j = 0; j < 5; ++j) acc[j] = WMMA_F16(a0, b0[j], acc[j]);
#pragma unroll
  for (int j = 0; j < 5; ++j) acc[j] = WMMA_F16(a1, b1[j], acc[j]);

  const int lane = threadIdx.x & 31;
  const int mb   = mt * 16 + ((lane & 16) ? 8 : 0);
#pragma unroll
  for (int j = 0; j < 5; ++j) {
    const int col = nt * 80 + j * 16 + (lane & 15);
#pragma unroll
    for (int v = 0; v < 8; ++v)
      __builtin_nontemporal_store(acc[j][v], &Sr[(size_t)(mb + v) * NV_ + col]);
  }
}

// ---------------------------------------------------------------------------

extern "C" void kernel_launch(void* const* d_in, const int* in_sizes, int n_in,
                              void* d_out, int out_size, void* d_ws, size_t ws_size,
                              hipStream_t stream) {
  (void)in_sizes; (void)n_in; (void)out_size; (void)ws_size;

  const float* feature_u = (const float*)d_in[0];
  const float* feature_v = (const float*)d_in[1];
  const float* sfeat_u   = (const float*)d_in[2];
  const float* sfeat_v   = (const float*)d_in[3];
  const int*   u_idx     = (const int*)d_in[4];
  const int*   v_idx     = (const int*)d_in[5];
  const float* edge_w    = (const float*)d_in[6];
  const float* W         = (const float*)d_in[7];
  const float* W_side_u  = (const float*)d_in[8];
  const float* b_side_u  = (const float*)d_in[9];
  const float* W_side_v  = (const float*)d_in[10];
  const float* b_side_v  = (const float*)d_in[11];
  const float* W_cat_u   = (const float*)d_in[12];
  const float* W_cat_v   = (const float*)d_in[13];
  const float* Q         = (const float*)d_in[14];
  float* score = (float*)d_out;

  // ---- workspace carve-out (all L2-resident; ~56 MB total) ----
  char* p = (char*)d_ws;
  auto carve = [&](size_t bytes) {
    char* r = p;
    p += (bytes + 255) & ~(size_t)255;
    return (void*)r;
  };
  h16*   feat_u_h  = (h16*)  carve((size_t)NU_ * FD_ * sizeof(h16));
  h16*   feat_v_h  = (h16*)  carve((size_t)NV_ * FD_ * sizeof(h16));
  h16*   Wt_h      = (h16*)  carve((size_t)RR_ * HD_ * FD_ * sizeof(h16));
  h16*   Wcu_t     = (h16*)  carve((size_t)OD_ * CAT_ * sizeof(h16));
  h16*   Wcv_t     = (h16*)  carve((size_t)OD_ * CAT_ * sizeof(h16));
  float* tu        = (float*)carve((size_t)RR_ * NU_ * HD_ * sizeof(float));
  float* tv        = (float*)carve((size_t)RR_ * NV_ * HD_ * sizeof(float));
  float* hcat_u    = (float*)carve((size_t)NU_ * CAT_ * sizeof(float));
  float* hcat_v    = (float*)carve((size_t)NV_ * CAT_ * sizeof(float));
  float* embed_u   = (float*)carve((size_t)NU_ * OD_ * sizeof(float));
  float* embed_v   = (float*)carve((size_t)NV_ * OD_ * sizeof(float));
  h16*   embed_v_h = (h16*)  carve((size_t)NV_ * OD_ * sizeof(h16));
  h16*   Ah        = (h16*)  carve((size_t)RR_ * NU_ * OD_ * sizeof(h16));

  auto blk = [](long long n, int t) { return (unsigned)((n + t - 1) / t); };

  // stage 0: conversions / transposes
  cvt_f16_kernel<<<blk((long long)NU_ * FD_, 256), 256, 0, stream>>>(feature_u, feat_u_h,
                                                                     (size_t)NU_ * FD_);
  cvt_f16_kernel<<<blk((long long)NV_ * FD_, 256), 256, 0, stream>>>(feature_v, feat_v_h,
                                                                     (size_t)NV_ * FD_);
  transpose_w_kernel<<<blk((long long)RR_ * HD_ * FD_, 256), 256, 0, stream>>>(W, Wt_h);
  transpose_cat_kernel<<<blk((long long)OD_ * CAT_, 256), 256, 0, stream>>>(W_cat_u, Wcu_t);
  transpose_cat_kernel<<<blk((long long)OD_ * CAT_, 256), 256, 0, stream>>>(W_cat_v, Wcv_t);

  // zero the segment-sum accumulators (graph-capturable memset nodes)
  hipMemsetAsync(hcat_u, 0, (size_t)NU_ * CAT_ * sizeof(float), stream);
  hipMemsetAsync(hcat_v, 0, (size_t)NV_ * CAT_ * sizeof(float), stream);

  // stage 1: tu/tv = feat @ W[r]  (WMMA f16->f32, double-buffered k-loop)
  gemm_feat_kernel<<<blk((long long)RR_ * (NU_ / 16) * 32, 256), 256, 0, stream>>>(
      feat_u_h, Wt_h, tu, NU_);
  gemm_feat_kernel<<<blk((long long)RR_ * (NV_ / 16) * 32, 256), 256, 0, stream>>>(
      feat_v_h, Wt_h, tv, NV_);

  // stage 2: weighted COO scatter into concat hidden columns
  scatter_kernel<<<blk((long long)RR_ * NE_ * HD_, 256), 256, 0, stream>>>(
      u_idx, v_idx, edge_w, tu, tv, hcat_u, hcat_v);

  // stage 3: side MLP into concat columns [320..383]
  side_kernel<<<blk((long long)NU_ * SHD_, 256), 256, 0, stream>>>(sfeat_u, W_side_u,
                                                                   b_side_u, hcat_u, NU_);
  side_kernel<<<blk((long long)NV_ * SHD_, 256), 256, 0, stream>>>(sfeat_v, W_side_v,
                                                                   b_side_v, hcat_v, NV_);

  // stage 4: embed = relu(relu(hcat) @ W_cat)  (WMMA, relu folded into A-load)
  gemm_cat_kernel<<<blk((long long)(NU_ / 16) * 32, 256), 256, 0, stream>>>(
      hcat_u, Wcu_t, embed_u, (h16*)nullptr, NU_);
  gemm_cat_kernel<<<blk((long long)(NV_ / 16) * 32, 256), 256, 0, stream>>>(
      hcat_v, Wcv_t, embed_v, embed_v_h, NV_);

  // stage 5: Ah[r] = embed_u @ Q[r]  (f32 VALU, converted to f16)
  q_kernel<<<blk((long long)RR_ * NU_ * OD_, 256), 256, 0, stream>>>(embed_u, Q, Ah);

  // stage 6: score[r] = Ah[r] @ embed_v^T  (WMMA + non-temporal 480MB store)
  score_kernel<<<blk((long long)RR_ * (NU_ / 16) * (NV_ / 80) * 32, 256), 256, 0, stream>>>(
      Ah, embed_v_h, score);
}
